// Transformer_18992345383487
// MI455X (gfx1250) — matrix operand order repaired and checked by reading, not compiled, not run on hardware
//
#include <hip/hip_runtime.h>

// ---------------- model constants (match reference) ----------------
#define cV 32000
#define cD 512
#define cH 8
#define cL 4
#define cF 2048
#define cB 4
#define cS 512
#define cT 512
#define cHD 64            // head dim = D/H
#define cBH (cB * cH)
#define cBS (cB * cS)     // 2048 rows encoder
#define cBT (cB * cT)     // 2048 rows decoder

// ---------------- GEMM tiling ----------------
#define BM 128            // block tile rows
#define BN 64             // block tile cols
#define BK 32             // k-step (one WMMA K)
#define NTHR 256          // 8 waves

typedef __attribute__((ext_vector_type(16))) _Float16 v16h;
typedef __attribute__((ext_vector_type(8)))  _Float16 v8h;
typedef __attribute__((ext_vector_type(8)))  float    v8f;

// ---------------- CDNA5 async-to-LDS helpers ----------------
typedef __attribute__((address_space(3))) void lds_void;

__device__ __forceinline__ unsigned lds_byte_off(const void* p) {
  // generic -> LDS(AS3) addrspacecast; integer value is the LDS byte offset
  return (unsigned)(unsigned long long)(lds_void*)p;
}

// one 16B async DMA: global -> LDS, tracked by ASYNCcnt (no VGPR data path)
__device__ __forceinline__ void async_ld_b128(const void* g, void* l) {
  unsigned loff = lds_byte_off(l);
  unsigned long long ga = (unsigned long long)g;
  asm volatile("global_load_async_to_lds_b128 %0, %1, off"
               :: "v"(loff), "v"(ga) : "memory");
}

__device__ __forceinline__ void wait_async0() {
#if __has_builtin(__builtin_amdgcn_s_wait_asynccnt)
  __builtin_amdgcn_s_wait_asynccnt(0);
#else
  asm volatile("s_wait_asynccnt 0x0" ::: "memory");
#endif
}

// =====================================================================
// Block-tiled WMMA engine (wave32, v_wmma_f32_16x16x32_f16):
//  block = 8 waves computing a 128x64 fp32 tile; each wave owns 32x32
//  (2x2 fragments): 4 WMMAs per k-step off 8 ds_load_b128 fragment loads.
//  Fragment layouts per ISA 7.12.2:
//    A frag: lane r=l&15 row, K = hf*8 + {0..7} and {16..23}   (hf=l>>4)
//    B frag: lane r col,     K = hf*8 + {0..7} and {16..23}
//    C/D:    col = r, VGPR j -> row j + 8*hf
// =====================================================================

// ---- shared: stage A tile (128x32 fp32 -> f16 in LDS, row-major) ----
__device__ __forceinline__ void stage_A_tile(const float* __restrict__ A, int lda,
                                             int kk, _Float16* __restrict__ sA) {
  const int t = threadIdx.x;
  const int row = t >> 1, cg = (t & 1) * 16;
  const float* p = A + (size_t)row * lda + kk + cg;
  v8h lo, hi;
#pragma unroll
  for (int i = 0; i < 8; ++i) { lo[i] = (_Float16)p[i]; hi[i] = (_Float16)p[i + 8]; }
  v8h* d = (v8h*)&sA[row * BK + cg];
  d[0] = lo; d[1] = hi;
}

// ---- shared: fragment loads + 4 WMMAs ----
__device__ __forceinline__ void frag_mma(const _Float16* __restrict__ sA,
                                         const _Float16* __restrict__ sBt,
                                         v8f acc[2][2]) {
  const int t = threadIdx.x, lane = t & 31, wave = t >> 5;
  const int hf = lane >> 4, r = lane & 15;
  const int wm = wave >> 1, wn = wave & 1;
  v16h afr[2], bfr[2];
#pragma unroll
  for (int am = 0; am < 2; ++am) {
    const int rr = wm * 32 + am * 16 + r;
    const v8h lo = *(const v8h*)&sA[rr * BK + hf * 8];
    const v8h hi = *(const v8h*)&sA[rr * BK + 16 + hf * 8];
#pragma unroll
    for (int e = 0; e < 8; ++e) { afr[am][e] = lo[e]; afr[am][e + 8] = hi[e]; }
  }
#pragma unroll
  for (int bn = 0; bn < 2; ++bn) {
    const int cc = wn * 32 + bn * 16 + r;
    const v8h lo = *(const v8h*)&sBt[cc * BK + hf * 8];
    const v8h hi = *(const v8h*)&sBt[cc * BK + 16 + hf * 8];
#pragma unroll
    for (int e = 0; e < 8; ++e) { bfr[bn][e] = lo[e]; bfr[bn][e + 8] = hi[e]; }
  }
#pragma unroll
  for (int am = 0; am < 2; ++am)
#pragma unroll
    for (int bn = 0; bn < 2; ++bn)
      acc[am][bn] = __builtin_amdgcn_wmma_f32_16x16x32_f16(
          false, afr[am], false, bfr[bn], (short)0, acc[am][bn], false, false);
}

// ---- fp32-B engine (attention: B converted inline). BTRANS=1: B is [N,K]. ----
template <int BTRANS>
__device__ __forceinline__ void gemm_tile_block(
    const float* __restrict__ A, int lda,
    const float* __restrict__ Bm, int ldb, int K,
    _Float16* __restrict__ sA, _Float16* __restrict__ sBt, v8f acc[2][2]) {
  const int t = threadIdx.x;
  for (int kk = 0; kk < K; kk += BK) {
    __syncthreads();
    stage_A_tile(A, lda, kk, sA);
    if (BTRANS) {
      const int col = t >> 2, kg = (t & 3) * 8;
      const float* p = Bm + (size_t)col * ldb + kk + kg;
      v8h h;
#pragma unroll
      for (int i = 0; i < 8; ++i) h[i] = (_Float16)p[i];
      *(v8h*)&sBt[col * BK + kg] = h;
    } else {
      const int k = t >> 3, cg = (t & 7) * 8;
      const float* p = Bm + (size_t)(kk + k) * ldb + cg;
#pragma unroll
      for (int i = 0; i < 8; ++i) sBt[(cg + i) * BK + k] = (_Float16)p[i];
    }
    __syncthreads();
    if (kk + BK < K) {
      __builtin_prefetch(A + (size_t)(t >> 1) * lda + kk + BK, 0, 1);
      if (BTRANS)
        __builtin_prefetch(Bm + (size_t)(t >> 2) * ldb + kk + BK, 0, 1);
      else
        __builtin_prefetch(Bm + (size_t)(kk + BK + (t & 31)) * ldb, 0, 1);
    }
    frag_mma(sA, sBt, acc);
  }
}

// ---- f16-weight engine: B tile DMA'd via global_load_async_to_lds_b128. ----
// Wt is the f16 weight arena stored [N][K] (K-contiguous), pre-offset to the
// block's column origin; one b128 async op per thread moves the 4KB B tile.
__device__ __forceinline__ void gemm_tile_block_async(
    const float* __restrict__ A, int lda,
    const _Float16* __restrict__ Wt, int K,
    _Float16* __restrict__ sA, _Float16* __restrict__ sBt, v8f acc[2][2]) {
  const int t = threadIdx.x;
  const int col = t >> 2, kg = (t & 3) * 8;
  for (int kk = 0; kk < K; kk += BK) {
    __syncthreads();
    // B tile: async DMA straight into LDS (no VGPR round-trip, ASYNCcnt)
    async_ld_b128(Wt + (size_t)col * K + kk + kg, &sBt[col * BK + kg]);
    // A tile: load fp32, cvt to f16, ds_store (overlaps with the DMA)
    stage_A_tile(A, lda, kk, sA);
    if (kk + BK < K)
      __builtin_prefetch(A + (size_t)(t >> 1) * lda + kk + BK, 0, 1);
    wait_async0();
    __syncthreads();
    frag_mma(sA, sBt, acc);
  }
}

// Epilogue: C/bias pre-offset to block origin; ldc = row stride.
__device__ __forceinline__ void gemm_epilogue(
    v8f acc[2][2], float* __restrict__ C, int ldc,
    const float* __restrict__ bias, int relu, float scale) {
  const int t = threadIdx.x, lane = t & 31, wave = t >> 5;
  const int hf = lane >> 4, r = lane & 15;
  const int wm = wave >> 1, wn = wave & 1;
#pragma unroll
  for (int bn = 0; bn < 2; ++bn) {
    const int col = wn * 32 + bn * 16 + r;
    const float bs = bias ? bias[col] : 0.0f;
#pragma unroll
    for (int am = 0; am < 2; ++am) {
#pragma unroll
      for (int j = 0; j < 8; ++j) {
        const int row = wm * 32 + am * 16 + j + hf * 8;
        float v = acc[am][bn][j] * scale + bs;
        if (relu) v = fmaxf(v, 0.0f);
        C[(size_t)row * ldc + col] = v;
      }
    }
  }
}

// ---------------- weight convert + transpose: W[K,N] f32 -> Wt[N,K] f16 -----
__global__ void k_cvtT(const float* __restrict__ W, _Float16* __restrict__ Wt,
                       int K, int N, int total) {
  int idx = blockIdx.x * blockDim.x + threadIdx.x;
  if (idx >= total) return;
  int k = idx / N, n = idx - k * N;
  Wt[(size_t)n * K + k] = (_Float16)W[idx];
}

// ---------------- weight GEMM: C[M,N] = A[M,K] @ W + bias (opt ReLU) --------
__global__ void __launch_bounds__(NTHR)
k_gemm(const float* __restrict__ A, const _Float16* __restrict__ Wt,
       const float* __restrict__ bias, float* __restrict__ C,
       int N, int K, int relu) {
  __shared__ _Float16 sA[BM * BK];
  __shared__ _Float16 sBt[BN * BK];
  const int bm0 = blockIdx.y * BM, bn0 = blockIdx.x * BN;
  v8f acc[2][2] = {};
  gemm_tile_block_async(A + (size_t)bm0 * K, K, Wt + (size_t)bn0 * K, K, sA, sBt, acc);
  gemm_epilogue(acc, C + (size_t)bm0 * N + bn0, N, bias ? bias + bn0 : nullptr, relu, 1.0f);
}

// ---------------- attention scores: sc[b,h,q,k] = (Q . K^T) * scale ----------
__global__ void __launch_bounds__(NTHR)
k_attn_scores(const float* __restrict__ q, const float* __restrict__ kmat,
              float* __restrict__ sc, int q_len, int k_len, float scale) {
  __shared__ _Float16 sA[BM * BK];
  __shared__ _Float16 sBt[BN * BK];
  const int z = blockIdx.z, b = z / cH, h = z % cH;
  const int bm0 = blockIdx.y * BM, bn0 = blockIdx.x * BN;
  const float* Ap = q    + (size_t)(b * q_len + bm0) * cD + h * cHD;
  const float* Bp = kmat + (size_t)(b * k_len + bn0) * cD + h * cHD;
  v8f acc[2][2] = {};
  gemm_tile_block<1>(Ap, cD, Bp, cD, cHD, sA, sBt, acc);
  float* Cp = sc + (size_t)z * q_len * k_len + (size_t)bm0 * k_len + bn0;
  gemm_epilogue(acc, Cp, k_len, nullptr, 0, scale);
}

// ---------------- attn @ V: out[b,q,h*64+d] = sum_k sc * v --------------------
__global__ void __launch_bounds__(NTHR)
k_attn_v(const float* __restrict__ sc, const float* __restrict__ v,
         float* __restrict__ out, int q_len, int k_len) {
  __shared__ _Float16 sA[BM * BK];
  __shared__ _Float16 sBt[BN * BK];
  const int z = blockIdx.z, b = z / cH, h = z % cH;
  const int bm0 = blockIdx.y * BM;     // N tile = full head (64), grid.x == 1
  const float* Ap = sc + (size_t)z * q_len * k_len + (size_t)bm0 * k_len;
  const float* Bp = v + (size_t)b * k_len * cD + h * cHD;
  v8f acc[2][2] = {};
  gemm_tile_block<0>(Ap, k_len, Bp, cD, k_len, sA, sBt, acc);
  float* Cp = out + (size_t)(b * q_len + bm0) * cD + h * cHD;
  gemm_epilogue(acc, Cp, cD, nullptr, 0, 1.0f);
}

// ---------------- embedding + sinusoidal positional encoding ----------------
__global__ void k_embed(const int* __restrict__ tok, const float* __restrict__ emb,
                        float* __restrict__ out, int seqlen, int total) {
  int idx = blockIdx.x * blockDim.x + threadIdx.x;
  if (idx >= total) return;
  int row = idx / cD;
  int d   = idx - row * cD;
  int s   = row % seqlen;
  int t   = tok[row];
  float de  = (float)(d & ~1);
  float ang = (float)s * __expf(de * (-9.210340371976184f / (float)cD));
  float pe  = (d & 1) ? __cosf(ang) : __sinf(ang);
  out[idx] = emb[(size_t)t * cD + d] * 22.627416997969522f /* sqrt(512) */ + pe;
}

// ---------------- masked softmax along k (pad mask, optional causal) --------
__global__ void k_mask_softmax(float* __restrict__ sc, const int* __restrict__ tok,
                               int q_len, int k_len, int causal) {
  const int qi = blockIdx.x, z = blockIdx.y, b = z / cH;
  float* row = sc + ((size_t)z * q_len + qi) * k_len;
  const int* tk = tok + (size_t)b * k_len;
  __shared__ float red[256];
  const int tid = threadIdx.x;
  float mx = -3.0e38f;
  for (int k = tid; k < k_len; k += 256) {
    bool m = (tk[k] != 0) && (!causal || (k <= qi));
    float e = m ? row[k] : -1.0e10f;
    row[k] = e;
    mx = fmaxf(mx, e);
  }
  red[tid] = mx; __syncthreads();
  for (int s = 128; s > 0; s >>= 1) { if (tid < s) red[tid] = fmaxf(red[tid], red[tid + s]); __syncthreads(); }
  mx = red[0]; __syncthreads();
  float sum = 0.0f;
  for (int k = tid; k < k_len; k += 256) { float e = __expf(row[k] - mx); row[k] = e; sum += e; }
  red[tid] = sum; __syncthreads();
  for (int s = 128; s > 0; s >>= 1) { if (tid < s) red[tid] += red[tid + s]; __syncthreads(); }
  const float inv = 1.0f / red[0];
  for (int k = tid; k < k_len; k += 256) row[k] *= inv;
}

// ---------------- fused residual add + LayerNorm (one block per row) --------
__global__ void k_add_ln(const float* __restrict__ x, const float* __restrict__ a,
                         const float* __restrict__ g, const float* __restrict__ bta,
                         float* __restrict__ out) {
  const int rowi = blockIdx.x;
  const float* xr = x + (size_t)rowi * cD;
  const float* ar = a + (size_t)rowi * cD;
  __shared__ float red[256];
  const int tid = threadIdx.x;
  float vals[2];
  float s0 = 0.0f;
#pragma unroll
  for (int i = 0; i < 2; ++i) { int d = tid + i * 256; vals[i] = xr[d] + ar[d]; s0 += vals[i]; }
  red[tid] = s0; __syncthreads();
  for (int s = 128; s > 0; s >>= 1) { if (tid < s) red[tid] += red[tid + s]; __syncthreads(); }
  const float mean = red[0] * (1.0f / cD); __syncthreads();
  float s1 = 0.0f;
#pragma unroll
  for (int i = 0; i < 2; ++i) { float t = vals[i] - mean; s1 += t * t; }
  red[tid] = s1; __syncthreads();
  for (int s = 128; s > 0; s >>= 1) { if (tid < s) red[tid] += red[tid + s]; __syncthreads(); }
  const float inv = rsqrtf(red[0] * (1.0f / cD) + 1e-5f);
#pragma unroll
  for (int i = 0; i < 2; ++i) {
    int d = tid + i * 256;
    out[(size_t)rowi * cD + d] = (vals[i] - mean) * inv * g[d] + bta[d];
  }
}

// =====================================================================
extern "C" void kernel_launch(void* const* d_in, const int* in_sizes, int n_in,
                              void* d_out, int out_size, void* d_ws, size_t ws_size,
                              hipStream_t stream) {
  (void)in_sizes; (void)n_in; (void)out_size; (void)ws_size;
  const int* src = (const int*)d_in[0];
  const int* trg = (const int*)d_in[1];
  auto Pf = [&](int i) { return (const float*)d_in[2 + i]; };
  // params (+2 into d_in):
  //  0 emb_src 1 emb_trg
  //  enc_sa: 2 Wq 3 bq 4 Wk 5 bk 6 Wv 7 bv 8 Wo 9 bo
  //  10 enc_W1 11 enc_b1 12 enc_W2 13 enc_b2
  //  14 enc_ln0_g 15 enc_ln0_b 16 enc_ln1_g 17 enc_ln1_b
  //  dec_sa: 18..25  dec_ca: 26..33
  //  34 dec_W1 35 dec_b1 36 dec_W2 37 dec_b2
  //  38..43 dec_ln{0,1,2}_{g,b}   44 W_out 45 b_out

  float* wsf = (float*)d_ws;
  const size_t ACT = (size_t)cBS * cD;
  float* xe     = wsf;
  float* xd     = xe + ACT;
  float* qb     = xd + ACT;
  float* kb     = qb + ACT;
  float* vb     = kb + ACT;
  float* attn   = vb + ACT;
  float* proj   = attn + ACT;
  float* ffn    = proj + ACT;                             // cBS * cF
  float* scores = ffn + (size_t)cBS * cF;                 // cB*cH*512*512
  _Float16* wtbuf = (_Float16*)(scores + (size_t)cBH * cS * cS);  // <= 512*32000 f16

  // weight GEMM: convert W[K,N]->Wt[N,K] f16 once, then async-DMA-fed WMMA
  auto gemm = [&](const float* A, const float* W, const float* bias, float* C,
                  int M, int N, int K, int relu) {
    const int tot = K * N;
    k_cvtT<<<(tot + 255) / 256, 256, 0, stream>>>(W, wtbuf, K, N, tot);
    dim3 g(N / BN, M / BM);
    k_gemm<<<g, NTHR, 0, stream>>>(A, wtbuf, bias, C, N, K, relu);
  };
  const float iscale = 0.125f; // 1/sqrt(64)

  // ---------------- encoder ----------------
  k_embed<<<(cBS * cD + 255) / 256, 256, 0, stream>>>(src, Pf(0), xe, cS, cBS * cD);
  for (int i = 0; i < cL; ++i) {
    const int DD = cD * cD;
    gemm(xe, Pf(2) + i * DD, Pf(3) + i * cD, qb, cBS, cD, cD, 0);
    gemm(xe, Pf(4) + i * DD, Pf(5) + i * cD, kb, cBS, cD, cD, 0);
    gemm(xe, Pf(6) + i * DD, Pf(7) + i * cD, vb, cBS, cD, cD, 0);
    k_attn_scores<<<dim3(cS / BN, cS / BM, cBH), NTHR, 0, stream>>>(qb, kb, scores, cS, cS, iscale);
    k_mask_softmax<<<dim3(cS, cBH), 256, 0, stream>>>(scores, src, cS, cS, 0);
    k_attn_v<<<dim3(1, cS / BM, cBH), NTHR, 0, stream>>>(scores, vb, attn, cS, cS);
    gemm(attn, Pf(8) + i * DD, Pf(9) + i * cD, proj, cBS, cD, cD, 0);
    k_add_ln<<<cBS, 256, 0, stream>>>(xe, proj, Pf(14) + i * cD, Pf(15) + i * cD, xe);
    gemm(xe, Pf(10) + i * cD * cF, Pf(11) + i * cF, ffn, cBS, cF, cD, 1);
    gemm(ffn, Pf(12) + i * cF * cD, Pf(13) + i * cD, proj, cBS, cD, cF, 0);
    k_add_ln<<<cBS, 256, 0, stream>>>(xe, proj, Pf(16) + i * cD, Pf(17) + i * cD, xe);
  }

  // ---------------- decoder ----------------
  k_embed<<<(cBT * cD + 255) / 256, 256, 0, stream>>>(trg, Pf(1), xd, cT, cBT * cD);
  for (int i = 0; i < cL; ++i) {
    const int DD = cD * cD;
    // masked self-attention
    gemm(xd, Pf(18) + i * DD, Pf(19) + i * cD, qb, cBT, cD, cD, 0);
    gemm(xd, Pf(20) + i * DD, Pf(21) + i * cD, kb, cBT, cD, cD, 0);
    gemm(xd, Pf(22) + i * DD, Pf(23) + i * cD, vb, cBT, cD, cD, 0);
    k_attn_scores<<<dim3(cT / BN, cT / BM, cBH), NTHR, 0, stream>>>(qb, kb, scores, cT, cT, iscale);
    k_mask_softmax<<<dim3(cT, cBH), 256, 0, stream>>>(scores, trg, cT, cT, 1);
    k_attn_v<<<dim3(1, cT / BM, cBH), NTHR, 0, stream>>>(scores, vb, attn, cT, cT);
    gemm(attn, Pf(24) + i * DD, Pf(25) + i * cD, proj, cBT, cD, cD, 0);
    k_add_ln<<<cBT, 256, 0, stream>>>(xd, proj, Pf(38) + i * cD, Pf(39) + i * cD, xd);
    // cross-attention (K/V from encoder output xe)
    gemm(xd, Pf(26) + i * DD, Pf(27) + i * cD, qb, cBT, cD, cD, 0);
    gemm(xe, Pf(28) + i * DD, Pf(29) + i * cD, kb, cBS, cD, cD, 0);
    gemm(xe, Pf(30) + i * DD, Pf(31) + i * cD, vb, cBS, cD, cD, 0);
    k_attn_scores<<<dim3(cS / BN, cT / BM, cBH), NTHR, 0, stream>>>(qb, kb, scores, cT, cS, iscale);
    k_mask_softmax<<<dim3(cT, cBH), 256, 0, stream>>>(scores, src, cT, cS, 0);
    k_attn_v<<<dim3(1, cT / BM, cBH), NTHR, 0, stream>>>(scores, vb, attn, cT, cS);
    gemm(attn, Pf(32) + i * DD, Pf(33) + i * cD, proj, cBT, cD, cD, 0);
    k_add_ln<<<cBT, 256, 0, stream>>>(xd, proj, Pf(40) + i * cD, Pf(41) + i * cD, xd);
    // FFN
    gemm(xd, Pf(34) + i * cD * cF, Pf(35) + i * cF, ffn, cBT, cF, cD, 1);
    gemm(ffn, Pf(36) + i * cF * cD, Pf(37) + i * cD, proj, cBT, cD, cF, 0);
    k_add_ln<<<cBT, 256, 0, stream>>>(xd, proj, Pf(42) + i * cD, Pf(43) + i * cD, xd);
  }

  // ---------------- output projection -> d_out [B,T,V] ----------------
  gemm(xd, Pf(44), Pf(45), (float*)d_out, cBT, cV, cD, 0);
}
